// MLADecoderSelfAttention_73787538145765
// MI455X (gfx1250) — matrix operand order repaired and checked by reading, not compiled, not run on hardware
//
#include <hip/hip_runtime.h>

// ---------------- CDNA5 WMMA types ----------------
typedef __attribute__((ext_vector_type(16))) __bf16 v16bf;
typedef __attribute__((ext_vector_type(8)))  float  v8f;

union Frag {               // 32 bytes = 16 bf16 = one 16-bit WMMA operand per lane
  uint4 q[2];
  v16bf v;
};

// ---------------- problem constants ----------------
#define S_LEN  4096
#define EMB    1024
#define NH     16
#define HDD    64
#define ROPE_D 32
#define NOPE_D 32
#define QR_D   512
#define KVR_D  512
#define KVDEC_W (NH * (NOPE_D + HDD))   // 1536
#define KVRAW_W (KVR_D + ROPE_D)        // 544

__device__ __forceinline__ unsigned short f2bf(float f) {
  unsigned u = __float_as_uint(f);
  u += 0x7FFFu + ((u >> 16) & 1u);     // round-to-nearest-even
  return (unsigned short)(u >> 16);
}

// gfx1250 async DMA: global -> LDS, one b128 per lane, tracked by ASYNCcnt
__device__ __forceinline__ void async_b128(unsigned lds_off, const void* gaddr) {
  asm volatile("global_load_async_to_lds_b128 %0, %1, off"
               :: "v"(lds_off), "v"(gaddr) : "memory");
}

// ---------------- fp32 -> bf16 convert ----------------
__global__ void cvt_f32_bf16(const float* __restrict__ in,
                             unsigned short* __restrict__ out, int n) {
  int i = blockIdx.x * blockDim.x + threadIdx.x;
  int stride = gridDim.x * blockDim.x;
  for (; i < n; i += stride) out[i] = f2bf(in[i]);
}

// ---------------- generic WMMA GEMM: C[M,N] = A[M,K] * W[N,K]^T ----------------
// 4 waves per block share one N-tile of W: the B tile (16 x 64 bf16 per K-chunk)
// is staged Global->LDS with async DMA (double buffered), read back via ds_load.
// A rows are unique per wave and stream straight from global.
// Requires: M % 64 == 0, N % 16 == 0, K % 64 == 0.
__global__ __launch_bounds__(128) void gemm_bf16_wmma(
    const unsigned short* __restrict__ A, const unsigned short* __restrict__ W,
    float* __restrict__ C, int M, int N, int K) {
  __shared__ __align__(16) unsigned short btile[2][16 * 64];  // 2 x 2KB
  int tid  = threadIdx.x;
  int lane = tid & 31;
  int wid  = tid >> 5;
  int hl   = lane >> 4;        // half-wave select
  int lm   = lane & 15;
  int tileN = blockIdx.x * 16;
  int tileM = (blockIdx.y * 4 + wid) * 16;

  // async-copy mapping: 128 threads x one b128 = one 16x64 bf16 B chunk
  int crow = tid >> 3;                 // 0..15: weight row within tile
  int cchk = tid & 7;                  // 0..7 : 8-element chunk within 64 K vals
  const unsigned short* wsrc = W + (size_t)(tileN + crow) * K + cchk * 8;
  unsigned lds0 = (unsigned)(uintptr_t)(&btile[0][0]) + crow * 128 + cchk * 16;
  unsigned lds1 = lds0 + 16 * 64 * 2;

  const unsigned short* arow = A + (size_t)(tileM + lm) * K;  // A: row m per lane
  v8f acc = {0.f, 0.f, 0.f, 0.f, 0.f, 0.f, 0.f, 0.f};

  async_b128(lds0, wsrc);              // stage K-chunk 0
  if (64 < K) async_b128(lds1, wsrc + 64);

  for (int k0 = 0; k0 < K; k0 += 64) {
    // current buffer resident when per-wave ASYNCcnt drops to #still-in-flight
    if (k0 + 64 < K) asm volatile("s_wait_asynccnt 0x1" ::: "memory");
    else             asm volatile("s_wait_asynccnt 0x0" ::: "memory");
    __syncthreads();

    const unsigned short* bbuf = &btile[(k0 >> 6) & 1][0];
    Frag a0, a1, b0, b1;
    // this code's A k order: lane half h holds K = 16h..16h+15 contiguous
    a0.q[0] = *(const uint4*)(arow + k0 + 16 * hl);
    a0.q[1] = *(const uint4*)(arow + k0 + 16 * hl + 8);
    a1.q[0] = *(const uint4*)(arow + k0 + 32 + 16 * hl);
    a1.q[1] = *(const uint4*)(arow + k0 + 32 + 16 * hl + 8);
    // this code's B k order (matches A): lane half h holds K = 16h..16h+15 contiguous (from LDS)
    b0.q[0] = *(const uint4*)(bbuf + lm * 64 + 16 * hl);
    b0.q[1] = *(const uint4*)(bbuf + lm * 64 + 16 * hl + 8);
    b1.q[0] = *(const uint4*)(bbuf + lm * 64 + 32 + 16 * hl);
    b1.q[1] = *(const uint4*)(bbuf + lm * 64 + 32 + 16 * hl + 8);
    __builtin_prefetch(arow + k0 + 128, 0, 0);
    acc = __builtin_amdgcn_wmma_f32_16x16x32_bf16(false, a0.v, false, b0.v,
                                                  (short)0, acc, false, false);
    acc = __builtin_amdgcn_wmma_f32_16x16x32_bf16(false, a1.v, false, b1.v,
                                                  (short)0, acc, false, false);
    __syncthreads();   // all waves done reading this buffer
    if (k0 + 128 < K) {
      unsigned dst = ((k0 >> 6) & 1) ? lds1 : lds0;   // chunk c+2 -> buffer (c)&1
      async_b128(dst, wsrc + k0 + 128);
    }
  }
  // C layout: vgpr r -> row tileM + r + 8*half, col tileN + lane%16
  float* crow_p = C + (size_t)(tileM + 8 * hl) * N + tileN + lm;
  for (int r = 0; r < 8; ++r) crow_p[(size_t)r * N] = acc[r];
}

// ---------------- LayerNorm (row-wise), bf16 output ----------------
__global__ __launch_bounds__(256) void layernorm_bf16(
    const float* __restrict__ X, int in_stride, const float* __restrict__ w,
    const float* __restrict__ b, unsigned short* __restrict__ out, int cols) {
  __shared__ float s_sum[256], s_sq[256];
  int row = blockIdx.x;
  const float* xr = X + (size_t)row * in_stride;
  float lsum = 0.f, lsq = 0.f;
  for (int c = threadIdx.x; c < cols; c += 256) {
    float v = xr[c]; lsum += v; lsq += v * v;
  }
  s_sum[threadIdx.x] = lsum; s_sq[threadIdx.x] = lsq;
  __syncthreads();
  for (int off = 128; off > 0; off >>= 1) {
    if ((int)threadIdx.x < off) {
      s_sum[threadIdx.x] += s_sum[threadIdx.x + off];
      s_sq[threadIdx.x]  += s_sq[threadIdx.x + off];
    }
    __syncthreads();
  }
  float mu  = s_sum[0] / (float)cols;
  float var = s_sq[0] / (float)cols - mu * mu;
  float inv = rsqrtf(var + 1e-5f);
  unsigned short* orow = out + (size_t)row * cols;
  for (int c = threadIdx.x; c < cols; c += 256)
    orow[c] = f2bf((xr[c] - mu) * inv * w[c] + b[c]);
}

// ---------------- build q_full (nope ++ roped pe), bf16 (S,H,64) ----------------
__global__ __launch_bounds__(64) void build_qfull(const float* __restrict__ q,
                                                  unsigned short* __restrict__ qf) {
  int s = blockIdx.x, h = blockIdx.y, d = threadIdx.x;
  const float* qr = q + ((size_t)s * NH + h) * HDD;
  float val;
  if (d < NOPE_D) {
    val = qr[d];
  } else {
    int p = d - NOPE_D, i = p >> 1;
    float x1 = qr[NOPE_D + 2 * i], x2 = qr[NOPE_D + 2 * i + 1];
    float invf = __powf(10000.f, -(float)(2 * i) / (float)ROPE_D);
    float ang = (float)s * invf;
    float c = __cosf(ang), sn = __sinf(ang);
    val = (p & 1) ? (x1 * sn + x2 * c) : (x1 * c - x2 * sn);
  }
  qf[((size_t)s * NH + h) * HDD + d] = f2bf(val);
}

// ---------------- build k_full (k_nope ++ broadcast roped k_pe), bf16 (S,H,64) ----
__global__ __launch_bounds__(64) void build_kfull(const float* __restrict__ kvdec,
                                                  const float* __restrict__ kvraw,
                                                  unsigned short* __restrict__ kf) {
  int s = blockIdx.x, h = blockIdx.y, d = threadIdx.x;
  float val;
  if (d < NOPE_D) {
    val = kvdec[(size_t)s * KVDEC_W + h * (NOPE_D + HDD) + d];
  } else {
    int p = d - NOPE_D, i = p >> 1;
    const float* kp = kvraw + (size_t)s * KVRAW_W + KVR_D;
    float x1 = kp[2 * i], x2 = kp[2 * i + 1];
    float invf = __powf(10000.f, -(float)(2 * i) / (float)ROPE_D);
    float ang = (float)s * invf;
    float c = __cosf(ang), sn = __sinf(ang);
    val = (p & 1) ? (x1 * sn + x2 * c) : (x1 * c - x2 * sn);
  }
  kf[((size_t)s * NH + h) * HDD + d] = f2bf(val);
}

// ---------------- build V^T: (H, 64, S) bf16 so P@V B-operand is K-contiguous ---
__global__ __launch_bounds__(256) void build_vT(const float* __restrict__ kvdec,
                                                unsigned short* __restrict__ vT) {
  int idx = blockIdx.x * blockDim.x + threadIdx.x;
  if (idx >= S_LEN * NH * HDD) return;
  int s = idx >> 10;
  int rem = idx & 1023;
  int h = rem >> 6, d = rem & 63;
  float val = kvdec[(size_t)s * KVDEC_W + h * (NOPE_D + HDD) + NOPE_D + d];
  vT[((size_t)(h * HDD + d)) * S_LEN + s] = f2bf(val);
}

// ---------------- flash attention: 1 wave per (16-row q-tile, head) ------------
__global__ __launch_bounds__(32) void mla_attention(
    const unsigned short* __restrict__ qf, const unsigned short* __restrict__ kf,
    const unsigned short* __restrict__ vT, unsigned short* __restrict__ out) {
  __shared__ __align__(16) unsigned short pbuf[16 * 32];  // P tile transpose bounce
  int lane = threadIdx.x;
  int hl = lane >> 4, lm = lane & 15;
  int qt = blockIdx.x, h = blockIdx.y;

  // Q A-fragments (16 rows x 64 feats -> two K=32 fragments), resident all loop
  const unsigned short* qrow = qf + ((size_t)(qt * 16 + lm) * NH + h) * HDD;
  Frag qa0, qa1;
  qa0.q[0] = *(const uint4*)(qrow + 16 * hl);
  qa0.q[1] = *(const uint4*)(qrow + 16 * hl + 8);
  qa1.q[0] = *(const uint4*)(qrow + 32 + 16 * hl);
  qa1.q[1] = *(const uint4*)(qrow + 32 + 16 * hl + 8);

  float m_i[8], l_i[8], alpha[8];
  for (int r = 0; r < 8; ++r) { m_i[r] = -1e30f; l_i[r] = 0.f; }
  v8f O0 = {0,0,0,0,0,0,0,0}, O1 = {0,0,0,0,0,0,0,0};
  v8f O2 = {0,0,0,0,0,0,0,0}, O3 = {0,0,0,0,0,0,0,0};
  const float scale = 0.125f;  // 1/sqrt(64)

  int jmax = (qt * 16 + 15) >> 5;           // causal: keys up to q row
  for (int j = 0; j <= jmax; ++j) {
    v8f s0 = {0,0,0,0,0,0,0,0}, s1 = {0,0,0,0,0,0,0,0};
    {  // key columns j*32 .. +15
      const unsigned short* kr = kf + ((size_t)(j * 32 + lm) * NH + h) * HDD + 16 * hl;
      Frag b0, b1;
      b0.q[0] = *(const uint4*)(kr);      b0.q[1] = *(const uint4*)(kr + 8);
      b1.q[0] = *(const uint4*)(kr + 32); b1.q[1] = *(const uint4*)(kr + 40);
      s0 = __builtin_amdgcn_wmma_f32_16x16x32_bf16(false, qa0.v, false, b0.v, (short)0, s0, false, false);
      s0 = __builtin_amdgcn_wmma_f32_16x16x32_bf16(false, qa1.v, false, b1.v, (short)0, s0, false, false);
    }
    {  // key columns j*32+16 .. +31
      const unsigned short* kr = kf + ((size_t)(j * 32 + 16 + lm) * NH + h) * HDD + 16 * hl;
      Frag b0, b1;
      b0.q[0] = *(const uint4*)(kr);      b0.q[1] = *(const uint4*)(kr + 8);
      b1.q[0] = *(const uint4*)(kr + 32); b1.q[1] = *(const uint4*)(kr + 40);
      s1 = __builtin_amdgcn_wmma_f32_16x16x32_bf16(false, qa0.v, false, b0.v, (short)0, s1, false, false);
      s1 = __builtin_amdgcn_wmma_f32_16x16x32_bf16(false, qa1.v, false, b1.v, (short)0, s1, false, false);
    }

    // online softmax over 32 columns; row m lives in (vgpr r, half hl),
    // columns spread over the 16 lanes of the half -> shfl_xor width 16
    for (int r = 0; r < 8; ++r) {
      int m = qt * 16 + r + 8 * hl;
      int n0 = j * 32 + lm;
      float v0 = s0[r] * scale; if (n0 > m)      v0 = -1e9f;
      float v1 = s1[r] * scale; if (n0 + 16 > m) v1 = -1e9f;
      float mx = fmaxf(v0, v1);
      for (int off = 8; off >= 1; off >>= 1) mx = fmaxf(mx, __shfl_xor(mx, off, 16));
      float mnew = fmaxf(m_i[r], mx);
      float a = __expf(m_i[r] - mnew);
      float e0 = __expf(v0 - mnew);
      float e1 = __expf(v1 - mnew);
      float rs = e0 + e1;
      for (int off = 8; off >= 1; off >>= 1) rs += __shfl_xor(rs, off, 16);
      l_i[r] = l_i[r] * a + rs;
      m_i[r] = mnew;
      alpha[r] = a;
      // stash P in LDS row-major (16 x 32) for C-layout -> A-layout transpose
      pbuf[(r + 8 * hl) * 32 + lm]      = f2bf(e0);
      pbuf[(r + 8 * hl) * 32 + 16 + lm] = f2bf(e1);
    }
    for (int r = 0; r < 8; ++r) {
      O0[r] *= alpha[r]; O1[r] *= alpha[r]; O2[r] *= alpha[r]; O3[r] *= alpha[r];
    }
    __syncthreads();

    // P as A-fragment (16x32)
    Frag pa;
    const unsigned short* pr = pbuf + lm * 32;
    pa.q[0] = *(const uint4*)(pr + 16 * hl);
    pa.q[1] = *(const uint4*)(pr + 16 * hl + 8);

    // O += P(16x32) @ V(32x64) : V^T rows are K-contiguous
    {
      const unsigned short* vb = vT + ((size_t)(h * HDD + lm)) * S_LEN + j * 32 + 16 * hl;
      Frag f; f.q[0] = *(const uint4*)(vb); f.q[1] = *(const uint4*)(vb + 8);
      O0 = __builtin_amdgcn_wmma_f32_16x16x32_bf16(false, pa.v, false, f.v, (short)0, O0, false, false);
    }
    {
      const unsigned short* vb = vT + ((size_t)(h * HDD + 16 + lm)) * S_LEN + j * 32 + 16 * hl;
      Frag f; f.q[0] = *(const uint4*)(vb); f.q[1] = *(const uint4*)(vb + 8);
      O1 = __builtin_amdgcn_wmma_f32_16x16x32_bf16(false, pa.v, false, f.v, (short)0, O1, false, false);
    }
    {
      const unsigned short* vb = vT + ((size_t)(h * HDD + 32 + lm)) * S_LEN + j * 32 + 16 * hl;
      Frag f; f.q[0] = *(const uint4*)(vb); f.q[1] = *(const uint4*)(vb + 8);
      O2 = __builtin_amdgcn_wmma_f32_16x16x32_bf16(false, pa.v, false, f.v, (short)0, O2, false, false);
    }
    {
      const unsigned short* vb = vT + ((size_t)(h * HDD + 48 + lm)) * S_LEN + j * 32 + 16 * hl;
      Frag f; f.q[0] = *(const uint4*)(vb); f.q[1] = *(const uint4*)(vb + 8);
      O3 = __builtin_amdgcn_wmma_f32_16x16x32_bf16(false, pa.v, false, f.v, (short)0, O3, false, false);
    }
    __syncthreads();   // protect pbuf before next iteration overwrites it
  }

  // finalize: out[(m)*1024 + h*64 + d], bf16
  for (int r = 0; r < 8; ++r) {
    float inv = 1.f / l_i[r];
    size_t m = (size_t)(qt * 16 + r + 8 * hl);
    unsigned short* orow = out + m * (NH * HDD) + h * HDD;
    orow[lm]      = f2bf(O0[r] * inv);
    orow[16 + lm] = f2bf(O1[r] * inv);
    orow[32 + lm] = f2bf(O2[r] * inv);
    orow[48 + lm] = f2bf(O3[r] * inv);
  }
}

// ================= host-side orchestration =================
extern "C" void kernel_launch(void* const* d_in, const int* in_sizes, int n_in,
                              void* d_out, int out_size, void* d_ws, size_t ws_size,
                              hipStream_t stream) {
  (void)in_sizes; (void)n_in; (void)out_size; (void)ws_size;
  const float* x     = (const float*)d_in[0];
  /* d_in[1] = mask: causal, applied analytically */
  const float* Wqa   = (const float*)d_in[2];
  const float* qa_w  = (const float*)d_in[3];
  const float* qa_b  = (const float*)d_in[4];
  const float* Wqb   = (const float*)d_in[5];
  const float* Wkva  = (const float*)d_in[6];
  const float* kva_w = (const float*)d_in[7];
  const float* kva_b = (const float*)d_in[8];
  const float* Wkvb  = (const float*)d_in[9];
  const float* Wo    = (const float*)d_in[10];
  float* out = (float*)d_out;

  char* ws = (char*)d_ws;
  size_t off = 0;
  auto alloc = [&](size_t bytes) -> void* {
    void* p = ws + off;
    off = (off + bytes + 255) & ~(size_t)255;
    return p;
  };

  unsigned short* xb    = (unsigned short*)alloc((size_t)S_LEN * EMB * 2);
  unsigned short* wqab  = (unsigned short*)alloc((size_t)QR_D * EMB * 2);
  unsigned short* wqbb  = (unsigned short*)alloc((size_t)(NH * HDD) * QR_D * 2);
  unsigned short* wkvab = (unsigned short*)alloc((size_t)KVRAW_W * EMB * 2);
  unsigned short* wkvbb = (unsigned short*)alloc((size_t)KVDEC_W * KVR_D * 2);
  unsigned short* wob   = (unsigned short*)alloc((size_t)EMB * (NH * HDD) * 2);
  float* qcraw = (float*)alloc((size_t)S_LEN * QR_D * 4);
  float* kvraw = (float*)alloc((size_t)S_LEN * KVRAW_W * 4);
  unsigned short* qcnb  = (unsigned short*)alloc((size_t)S_LEN * QR_D * 2);
  unsigned short* ckvnb = (unsigned short*)alloc((size_t)S_LEN * KVR_D * 2);
  float* qf32  = (float*)alloc((size_t)S_LEN * NH * HDD * 4);
  float* kvdec = (float*)alloc((size_t)S_LEN * KVDEC_W * 4);
  unsigned short* qfull = (unsigned short*)alloc((size_t)S_LEN * NH * HDD * 2);
  unsigned short* kfull = (unsigned short*)alloc((size_t)S_LEN * NH * HDD * 2);
  unsigned short* vTb   = (unsigned short*)alloc((size_t)NH * HDD * S_LEN * 2);
  unsigned short* attb  = (unsigned short*)alloc((size_t)S_LEN * NH * HDD * 2);

  dim3 cb(256);
  cvt_f32_bf16<<<1024, cb, 0, stream>>>(x,    xb,    S_LEN * EMB);
  cvt_f32_bf16<<<512,  cb, 0, stream>>>(Wqa,  wqab,  QR_D * EMB);
  cvt_f32_bf16<<<512,  cb, 0, stream>>>(Wqb,  wqbb,  NH * HDD * QR_D);
  cvt_f32_bf16<<<512,  cb, 0, stream>>>(Wkva, wkvab, KVRAW_W * EMB);
  cvt_f32_bf16<<<512,  cb, 0, stream>>>(Wkvb, wkvbb, KVDEC_W * KVR_D);
  cvt_f32_bf16<<<1024, cb, 0, stream>>>(Wo,   wob,   EMB * NH * HDD);

  // q_c_raw = x @ Wqa^T   (4096 x 512, K=1024)
  gemm_bf16_wmma<<<dim3(QR_D / 16, S_LEN / 64), 128, 0, stream>>>(
      xb, wqab, qcraw, S_LEN, QR_D, EMB);
  // kv_raw = x @ Wkva^T   (4096 x 544, K=1024)
  gemm_bf16_wmma<<<dim3(KVRAW_W / 16, S_LEN / 64), 128, 0, stream>>>(
      xb, wkvab, kvraw, S_LEN, KVRAW_W, EMB);

  layernorm_bf16<<<S_LEN, 256, 0, stream>>>(qcraw, QR_D,    qa_w,  qa_b,  qcnb,  QR_D);
  layernorm_bf16<<<S_LEN, 256, 0, stream>>>(kvraw, KVRAW_W, kva_w, kva_b, ckvnb, KVR_D);

  // q = q_cn @ Wqb^T      (4096 x 1024, K=512)
  gemm_bf16_wmma<<<dim3((NH * HDD) / 16, S_LEN / 64), 128, 0, stream>>>(
      qcnb, wqbb, qf32, S_LEN, NH * HDD, QR_D);
  // kv_dec = ckv_n @ Wkvb^T (4096 x 1536, K=512)
  gemm_bf16_wmma<<<dim3(KVDEC_W / 16, S_LEN / 64), 128, 0, stream>>>(
      ckvnb, wkvbb, kvdec, S_LEN, KVDEC_W, KVR_D);

  build_qfull<<<dim3(S_LEN, NH), 64, 0, stream>>>(qf32, qfull);
  build_kfull<<<dim3(S_LEN, NH), 64, 0, stream>>>(kvdec, kvraw, kfull);
  build_vT<<<(S_LEN * NH * HDD + 255) / 256, 256, 0, stream>>>(kvdec, vTb);

  mla_attention<<<dim3(S_LEN / 16, NH), 32, 0, stream>>>(qfull, kfull, vTb, attb);

  // final = attn @ Wo^T   (4096 x 1024, K=1024) -> fp32 output
  gemm_bf16_wmma<<<dim3(EMB / 16, S_LEN / 64), 128, 0, stream>>>(
      attb, wob, out, S_LEN, EMB, NH * HDD);
}
